// LlamaAttention_40364102648358
// MI455X (gfx1250) — matrix operand order repaired and checked by reading, not compiled, not run on hardware
//
#include <hip/hip_runtime.h>
#include <hip/hip_bf16.h>

typedef __attribute__((ext_vector_type(16))) __bf16 v16bf;
typedef __attribute__((ext_vector_type(8)))  __bf16 v8bf;
typedef __attribute__((ext_vector_type(8)))  float  v8f;

#define DIM   4096
#define NH    32
#define NKV   8
#define HDIM  128
#define NQKV  6144   // 4096 (q) + 1024 (k) + 1024 (v)

// ---------------------------------------------------------------------------
// WMMA fragment helpers (CDNA5 ISA 7.12.2, 16-bit operands, wave32)
// A 16x32:  lane m = lane&15, kh = lane>>4 ; f[0..7] = K kh*8+0..7,
//           f[8..15] = K 16+kh*8+0..7  -> two 16B chunks at +0 and +16 elems.
// B 32x16:  lane n = lane&15, kh = lane>>4 ; same K order as A (chunks at +0, +16).
// C/D f32:  row = r + 8*kh, col = lane&15.
// ---------------------------------------------------------------------------
__device__ __forceinline__ v16bf frag_a(const __bf16* p) {   // p = rowbase + k0 + kh*8
    v8bf lo = *(const v8bf*)p;
    v8bf hi = *(const v8bf*)(p + 16);
    return __builtin_shufflevector(lo, hi, 0,1,2,3,4,5,6,7,8,9,10,11,12,13,14,15);
}
__device__ __forceinline__ v16bf frag_b(const __bf16* p) {   // p = rowbase + k0 + kh*8
    v8bf lo = *(const v8bf*)p;
    v8bf hi = *(const v8bf*)(p + 16);
    return __builtin_shufflevector(lo, hi, 0,1,2,3,4,5,6,7,8,9,10,11,12,13,14,15);
}
__device__ __forceinline__ v8f wmma_bf16(v16bf a, v16bf b, v8f c) {
    return __builtin_amdgcn_wmma_f32_16x16x32_bf16(false, a, false, b, (short)0, c, false, false);
}

// CDNA5 async global->LDS copy (16B per lane), tracked by ASYNCcnt.
// Low 32 bits of a generic LDS pointer are the LDS byte offset (ISA 10.2).
__device__ __forceinline__ void async_load_b128(const void* gptr, void* lptr) {
    unsigned int lds = (unsigned int)(unsigned long long)lptr;
    asm volatile("global_load_async_to_lds_b128 %0, %1, off"
                 :: "v"(lds), "v"((unsigned long long)gptr)
                 : "memory");
}
__device__ __forceinline__ void wait_asynccnt0() {
    asm volatile("s_wait_asynccnt 0x0" ::: "memory");
}

// ---------------------------------------------------------------------------
// RMSNorm: one block per row, output bf16
// ---------------------------------------------------------------------------
__global__ __launch_bounds__(256)
void rmsnorm_kernel(const float* __restrict__ x, const float* __restrict__ w,
                    __bf16* __restrict__ out) {
    __shared__ float red[8];
    const int s = blockIdx.x, t = threadIdx.x;
    const float* row = x + (size_t)s * DIM;
    float sum = 0.f;
#pragma unroll
    for (int i = 0; i < DIM / 256; ++i) {
        float v = row[t + i * 256];
        sum += v * v;
    }
#pragma unroll
    for (int off = 16; off >= 1; off >>= 1) sum += __shfl_xor(sum, off, 32);
    if ((t & 31) == 0) red[t >> 5] = sum;
    __syncthreads();
    if (t == 0) {
        float tot = 0.f;
#pragma unroll
        for (int i = 0; i < 8; ++i) tot += red[i];
        red[0] = tot;
    }
    __syncthreads();
    const float scale = rsqrtf(red[0] * (1.0f / DIM) + 1e-5f);
#pragma unroll
    for (int i = 0; i < DIM / 256; ++i) {
        int d = t + i * 256;
        out[(size_t)s * DIM + d] = (__bf16)(row[d] * scale * w[d]);
    }
}

// ---------------------------------------------------------------------------
// Transpose fp32 [K][N] -> bf16 [N][outStride], out[n*outStride + k]
// ---------------------------------------------------------------------------
__global__ void transpose_bf16_kernel(const float* __restrict__ in, __bf16* __restrict__ out,
                                      int K, int N, int outStride) {
    __shared__ float tile[32][33];
    const int n0 = blockIdx.x * 32, k0 = blockIdx.y * 32;
    const int tx = threadIdx.x, ty = threadIdx.y;
#pragma unroll
    for (int i = 0; i < 4; ++i)
        tile[ty + i * 8][tx] = in[(size_t)(k0 + ty + i * 8) * N + n0 + tx];
    __syncthreads();
#pragma unroll
    for (int i = 0; i < 4; ++i)
        out[(size_t)(n0 + ty + i * 8) * outStride + k0 + tx] = (__bf16)tile[tx][ty + i * 8];
}

// ---------------------------------------------------------------------------
// WMMA GEMM: C[M][N] = A[M][K] * Bt[N][K]^T (+ add). Wave computes 16x64.
// K pipeline unrolled by 2 with alternating fragment buffers (no inter-buffer
// register copies): each buffer is reloaded in place right after its WMMA
// group consumes it. Requires K % 64 == 0 and K >= 64 (here K = 4096).
// Prefetch pulls ~2KB ahead (global_prefetch_b8, immediate offsets).
// Block = 256 threads = 8 waves -> 16 x 512 tile. grid = (M/16, N/512).
// ---------------------------------------------------------------------------
__global__ __launch_bounds__(256)
void gemm_wmma_kernel(const __bf16* __restrict__ A, const __bf16* __restrict__ Bt,
                      float* __restrict__ C, const float* __restrict__ addp,
                      int M, int N, int K) {
    const int w = threadIdx.x >> 5, lane = threadIdx.x & 31;
    const int mrow = lane & 15, kh = lane >> 4;
    const int m0 = blockIdx.x * 16;
    const int n0 = blockIdx.y * 512 + w * 64;

    v8f acc[4];
#pragma unroll
    for (int t = 0; t < 4; ++t) acc[t] = (v8f){0.f,0.f,0.f,0.f,0.f,0.f,0.f,0.f};

    const __bf16* Ap = A + (size_t)(m0 + mrow) * K + kh * 8;
    const __bf16* Bp[4];
#pragma unroll
    for (int t = 0; t < 4; ++t)
        Bp[t] = Bt + (size_t)(n0 + t * 16 + mrow) * K + kh * 8;

    v16bf a0, a1, b0[4], b1[4];
    // steps of 32 in K; nSteps = K/32 (even)
    a0 = frag_a(Ap);
#pragma unroll
    for (int t = 0; t < 4; ++t) b0[t] = frag_b(Bp[t]);
    a1 = frag_a(Ap + 32);
#pragma unroll
    for (int t = 0; t < 4; ++t) b1[t] = frag_b(Bp[t] + 32);

    for (int k0 = 0; k0 < K - 64; k0 += 64) {
        __builtin_prefetch(Ap + k0 + 1056, 0, 1);
#pragma unroll
        for (int t = 0; t < 4; ++t) __builtin_prefetch(Bp[t] + k0 + 1056, 0, 1);

        // consume buffer 0 (step k0), then refill it with step k0+64
#pragma unroll
        for (int t = 0; t < 4; ++t) acc[t] = wmma_bf16(a0, b0[t], acc[t]);
        a0 = frag_a(Ap + k0 + 64);
#pragma unroll
        for (int t = 0; t < 4; ++t) b0[t] = frag_b(Bp[t] + k0 + 64);

        // consume buffer 1 (step k0+32), then refill with step k0+96
#pragma unroll
        for (int t = 0; t < 4; ++t) acc[t] = wmma_bf16(a1, b1[t], acc[t]);
        a1 = frag_a(Ap + k0 + 96);
#pragma unroll
        for (int t = 0; t < 4; ++t) b1[t] = frag_b(Bp[t] + k0 + 96);
    }
    // epilogue: last two steps already resident in the buffers
#pragma unroll
    for (int t = 0; t < 4; ++t) acc[t] = wmma_bf16(a0, b0[t], acc[t]);
#pragma unroll
    for (int t = 0; t < 4; ++t) acc[t] = wmma_bf16(a1, b1[t], acc[t]);

#pragma unroll
    for (int t = 0; t < 4; ++t) {
#pragma unroll
        for (int r = 0; r < 8; ++r) {
            int m = m0 + r + 8 * kh;
            int n = n0 + t * 16 + mrow;
            size_t idx = (size_t)m * N + n;
            float v = acc[t][r];
            if (addp) v += addp[idx];
            C[idx] = v;
        }
    }
}

// ---------------------------------------------------------------------------
// RoPE (interleaved) + split qkv -> q_bf16 (scaled by 1/sqrt(HD)),
// k_bf16/v_bf16, and fp32 k/v into the output caches at position_ids[s].
// grid = (S, NQKV/512), block = 256 ; each thread handles one even/odd pair.
// ---------------------------------------------------------------------------
__global__ __launch_bounds__(256)
void rope_split_kernel(const float* __restrict__ qkv,
                       const float* __restrict__ cosp, const float* __restrict__ sinp,
                       const int* __restrict__ pid,
                       float* __restrict__ outK, float* __restrict__ outV,
                       __bf16* __restrict__ qb, __bf16* __restrict__ kb,
                       __bf16* __restrict__ vb) {
    const int s = blockIdx.x;
    const int p = blockIdx.y * 256 + threadIdx.x;   // pair index, 0..3071
    const int e0 = 2 * p;
    const float x0 = qkv[(size_t)s * NQKV + e0];
    const float x1 = qkv[(size_t)s * NQKV + e0 + 1];
    const int pos = pid[s];
    const float QSCALE = 0.08838834764831845f;      // 1/sqrt(128)

    if (e0 < DIM) {                                 // Q
        int h = e0 >> 7, d = e0 & 127;
        float c = cosp[(size_t)s * HDIM + d], si = sinp[(size_t)s * HDIM + d];
        float r0 = x0 * c - x1 * si;
        float r1 = x1 * c + x0 * si;
        size_t o = ((size_t)s * NH + h) * HDIM + d;
        qb[o]     = (__bf16)(r0 * QSCALE);
        qb[o + 1] = (__bf16)(r1 * QSCALE);
    } else if (e0 < DIM + NKV * HDIM) {             // K (rope + cache)
        int idx = e0 - DIM, kvh = idx >> 7, d = idx & 127;
        float c = cosp[(size_t)s * HDIM + d], si = sinp[(size_t)s * HDIM + d];
        float r0 = x0 * c - x1 * si;
        float r1 = x1 * c + x0 * si;
        size_t oc = ((size_t)pos * NKV + kvh) * HDIM + d;
        outK[oc] = r0; outK[oc + 1] = r1;
        size_t ob = ((size_t)s * NKV + kvh) * HDIM + d;
        kb[ob] = (__bf16)r0; kb[ob + 1] = (__bf16)r1;
    } else {                                        // V (no rope)
        int idx = e0 - DIM - NKV * HDIM, kvh = idx >> 7, d = idx & 127;
        size_t oc = ((size_t)pos * NKV + kvh) * HDIM + d;
        outV[oc] = x0; outV[oc + 1] = x1;
        size_t ob = ((size_t)s * NKV + kvh) * HDIM + d;
        vb[ob] = (__bf16)x0; vb[ob + 1] = (__bf16)x1;
    }
}

// ---------------------------------------------------------------------------
// Causal GQA flash attention. grid = (S/128, NH), block = 256 (8 waves).
// Wave w owns q rows [blk*128 + w*16, +16). Key tiles of 32.
// Double-buffered LDS: K tile staged via global_load_async_to_lds_b128
// (ASYNCcnt) for tile j+1 while tile j computes; V tile loaded to registers
// during staging and scattered (transposed, stride 40) after the S phase.
// P converts C-layout -> A-layout via per-wave LDS slab; only wave-local
// ordering needed (LDS in-order per wave) -> wave_barrier, one hard barrier
// per tile. Output bf16 [S][NH*HD].
// ---------------------------------------------------------------------------
__global__ __launch_bounds__(256)
void attn_kernel(const __bf16* __restrict__ qb, const __bf16* __restrict__ kb,
                 const __bf16* __restrict__ vb, __bf16* __restrict__ attn) {
    __shared__ __align__(16) __bf16 Klds[2][32 * 128];     // [buf][key][d]
    __shared__ __align__(16) __bf16 Vt[2][128 * 40];       // [buf][d][key], padded
    __shared__ __align__(16) __bf16 Plds[8 * 16 * 32];     // per-wave [qrow][key]

    const int head = blockIdx.y, kvh = head >> 2;
    const int qRow0 = blockIdx.x * 128;
    const int tid = threadIdx.x;
    const int w = tid >> 5, lane = tid & 31;
    const int mrow = lane & 15, kh = lane >> 4;
    const int qWave = qRow0 + w * 16;

    // staging role of this thread
    const int si = tid >> 3;              // key 0..31
    const int sd0 = (tid & 7) * 16;       // d chunk base

    // Q fragments for this wave's 16 rows (pre-scaled by 1/sqrt(HD))
    const __bf16* qptr = qb + ((size_t)(qWave + mrow) * NH + head) * HDIM;
    v16bf qf[4];
#pragma unroll
    for (int c = 0; c < 4; ++c) qf[c] = frag_a(qptr + c * 32 + kh * 8);

    v8f o[8];
#pragma unroll
    for (int c = 0; c < 8; ++c) o[c] = (v8f){0.f,0.f,0.f,0.f,0.f,0.f,0.f,0.f};
    float mrun[8], lrun[8];
#pragma unroll
    for (int r = 0; r < 8; ++r) { mrun[r] = -1e30f; lrun[r] = 0.f; }

    const int nTiles = (qRow0 + 128) / 32;

    // ---- prologue: stage tile 0 into buffer 0 ----
    {
        const __bf16* ksrc = kb + ((size_t)si * NKV + kvh) * HDIM + sd0;
        async_load_b128(ksrc,     &Klds[0][si * 128 + sd0]);
        async_load_b128(ksrc + 8, &Klds[0][si * 128 + sd0 + 8]);
        const __bf16* vsrc = vb + ((size_t)si * NKV + kvh) * HDIM + sd0;
        v8bf a0 = *(const v8bf*)vsrc;
        v8bf a1 = *(const v8bf*)(vsrc + 8);
#pragma unroll
        for (int u = 0; u < 8; ++u) Vt[0][(sd0 + u) * 40 + si] = a0[u];
#pragma unroll
        for (int u = 0; u < 8; ++u) Vt[0][(sd0 + 8 + u) * 40 + si] = a1[u];
    }

    for (int j = 0; j < nTiles; ++j) {
        wait_asynccnt0();
        __syncthreads();
        const int cur = j & 1, nxt = cur ^ 1;
        const int key0 = j * 32;
        const bool haveNext = (j + 1) < nTiles;

        // ---- issue next-tile staging: async K, V into registers ----
        v8bf va0 = {}, va1 = {};
        if (haveNext) {
            const __bf16* ksrc = kb + ((size_t)(key0 + 32 + si) * NKV + kvh) * HDIM + sd0;
            async_load_b128(ksrc,     &Klds[nxt][si * 128 + sd0]);
            async_load_b128(ksrc + 8, &Klds[nxt][si * 128 + sd0 + 8]);
            const __bf16* vsrc = vb + ((size_t)(key0 + 32 + si) * NKV + kvh) * HDIM + sd0;
            va0 = *(const v8bf*)vsrc;
            va1 = *(const v8bf*)(vsrc + 8);
        }

        const bool active = key0 <= qWave + 15;
        if (active) {
            // ---- S = Q K^T for 32 keys (two 16-col fragments) ----
            v8f s0 = (v8f){0.f,0.f,0.f,0.f,0.f,0.f,0.f,0.f};
            v8f s1 = (v8f){0.f,0.f,0.f,0.f,0.f,0.f,0.f,0.f};
#pragma unroll
            for (int c = 0; c < 4; ++c) {
                v16bf b0 = frag_b(&Klds[cur][(mrow) * 128 + c * 32 + kh * 8]);
                s0 = wmma_bf16(qf[c], b0, s0);
            }
#pragma unroll
            for (int c = 0; c < 4; ++c) {
                v16bf b1 = frag_b(&Klds[cur][(16 + mrow) * 128 + c * 32 + kh * 8]);
                s1 = wmma_bf16(qf[c], b1, s1);
            }
            // ---- online softmax (row = r + 8*kh, col = mrow) ----
            const int kc0 = key0 + mrow, kc1 = key0 + 16 + mrow;
#pragma unroll
            for (int r = 0; r < 8; ++r) {
                const int qg = qWave + r + 8 * kh;
                const bool ok0 = (kc0 <= qg), ok1 = (kc1 <= qg);
                float tm = fmaxf(ok0 ? s0[r] : -1e30f, ok1 ? s1[r] : -1e30f);
#pragma unroll
                for (int off = 1; off <= 8; off <<= 1) tm = fmaxf(tm, __shfl_xor(tm, off, 32));
                const float mnew = fmaxf(mrun[r], tm);
                const float corr = __expf(mrun[r] - mnew);
                const float p0 = ok0 ? __expf(s0[r] - mnew) : 0.f;
                const float p1 = ok1 ? __expf(s1[r] - mnew) : 0.f;
                float rs = p0 + p1;
#pragma unroll
                for (int off = 1; off <= 8; off <<= 1) rs += __shfl_xor(rs, off, 32);
                lrun[r] = lrun[r] * corr + rs;
                mrun[r] = mnew;
#pragma unroll
                for (int c = 0; c < 8; ++c) o[c][r] *= corr;
                __bf16* prow = &Plds[(w * 16 + r + 8 * kh) * 32];
                prow[mrow]      = (__bf16)p0;
                prow[16 + mrow] = (__bf16)p1;
            }
        }

        // ---- scatter next-tile V (transposed) while S-latency drains ----
        if (haveNext) {
#pragma unroll
            for (int u = 0; u < 8; ++u) Vt[nxt][(sd0 + u) * 40 + si] = va0[u];
#pragma unroll
            for (int u = 0; u < 8; ++u) Vt[nxt][(sd0 + 8 + u) * 40 + si] = va1[u];
        }

        // P write -> P read is wave-local (LDS in-order per wave)
        __builtin_amdgcn_wave_barrier();
        asm volatile("" ::: "memory");

        if (active) {
            // ---- O += P V : P as A-fragment (16x32), V^T as B-fragments ----
            v16bf pf = frag_a(&Plds[(w * 16 + mrow) * 32 + kh * 8]);
#pragma unroll
            for (int c = 0; c < 8; ++c) {
                v16bf vf = frag_b(&Vt[cur][(c * 16 + mrow) * 40 + kh * 8]);
                o[c] = wmma_bf16(pf, vf, o[c]);
            }
        }
    }

    // ---- epilogue: O /= l, store bf16 [s][head*HD + d] ----
#pragma unroll
    for (int r = 0; r < 8; ++r) {
        const float inv = 1.f / lrun[r];
        const size_t row = (size_t)(qWave + r + 8 * kh) * (NH * HDIM) + head * HDIM;
#pragma unroll
        for (int c = 0; c < 8; ++c)
            attn[row + c * 16 + mrow] = (__bf16)(o[c][r] * inv);
    }
}

// ---------------------------------------------------------------------------
extern "C" void kernel_launch(void* const* d_in, const int* in_sizes, int n_in,
                              void* d_out, int out_size, void* d_ws, size_t ws_size,
                              hipStream_t stream) {
    (void)n_in; (void)out_size; (void)ws_size;
    const float* hs   = (const float*)d_in[0];
    const float* cosp = (const float*)d_in[1];
    const float* sinp = (const float*)d_in[2];
    const float* kc   = (const float*)d_in[3];
    const float* vc   = (const float*)d_in[4];
    const float* Wq   = (const float*)d_in[5];
    const float* Wk   = (const float*)d_in[6];
    const float* Wv   = (const float*)d_in[7];
    const float* Wo   = (const float*)d_in[8];
    const float* rw   = (const float*)d_in[9];
    const int*   pid  = (const int*)d_in[10];

    const int S      = in_sizes[0] / DIM;           // 2048
    const int MAXSEQ = in_sizes[3] / (NKV * HDIM);  // 4096

    float* outWR = (float*)d_out;
    float* outK  = outWR + (size_t)S * DIM;
    float* outV  = outK + (size_t)MAXSEQ * NKV * HDIM;

    char* ws = (char*)d_ws;
    auto alloc = [&](size_t bytes) {
        char* p = ws; ws += (bytes + 255) & ~(size_t)255; return p;
    };
    __bf16* h_bf  = (__bf16*)alloc((size_t)S * DIM * 2);
    __bf16* WtQKV = (__bf16*)alloc((size_t)NQKV * DIM * 2);
    float*  qkv   = (float*) alloc((size_t)S * NQKV * 4);
    __bf16* q_bf  = (__bf16*)alloc((size_t)S * NH * HDIM * 2);
    __bf16* k_bf  = (__bf16*)alloc((size_t)S * NKV * HDIM * 2);
    __bf16* v_bf  = (__bf16*)alloc((size_t)S * NKV * HDIM * 2);
    __bf16* at_bf = (__bf16*)alloc((size_t)S * NH * HDIM * 2);
    __bf16* WoT   = (__bf16*)alloc((size_t)DIM * DIM * 2);

    // caches: copy input cache first, rope kernel then scatters rows [0,S)
    hipMemcpyAsync(outK, kc, (size_t)MAXSEQ * NKV * HDIM * 4, hipMemcpyDeviceToDevice, stream);
    hipMemcpyAsync(outV, vc, (size_t)MAXSEQ * NKV * HDIM * 4, hipMemcpyDeviceToDevice, stream);

    rmsnorm_kernel<<<S, 256, 0, stream>>>(hs, rw, h_bf);

    dim3 tb(32, 8);
    transpose_bf16_kernel<<<dim3(DIM / 32, DIM / 32), tb, 0, stream>>>(Wq, WtQKV, DIM, DIM, DIM);
    transpose_bf16_kernel<<<dim3((NKV * HDIM) / 32, DIM / 32), tb, 0, stream>>>(
        Wk, WtQKV + (size_t)DIM * DIM, DIM, NKV * HDIM, DIM);
    transpose_bf16_kernel<<<dim3((NKV * HDIM) / 32, DIM / 32), tb, 0, stream>>>(
        Wv, WtQKV + (size_t)(DIM + NKV * HDIM) * DIM, DIM, NKV * HDIM, DIM);
    transpose_bf16_kernel<<<dim3(DIM / 32, DIM / 32), tb, 0, stream>>>(Wo, WoT, DIM, DIM, DIM);

    // fused QKV projection: [S,4096] x [4096,6144]
    gemm_wmma_kernel<<<dim3(S / 16, NQKV / 512), 256, 0, stream>>>(
        h_bf, WtQKV, qkv, nullptr, S, NQKV, DIM);

    rope_split_kernel<<<dim3(S, (NQKV / 2) / 256), 256, 0, stream>>>(
        qkv, cosp, sinp, pid, outK, outV, q_bf, k_bf, v_bf);

    attn_kernel<<<dim3(S / 128, NH), 256, 0, stream>>>(q_bf, k_bf, v_bf, at_bf);

    // output projection + residual, straight into d_out
    gemm_wmma_kernel<<<dim3(S / 16, DIM / 512), 256, 0, stream>>>(
        at_bf, WoT, outWR, hs, S, DIM, DIM);
}
